// LSTM_OUT_44822278701285
// MI455X (gfx1250) — matrix operand order repaired and checked by reading, not compiled, not run on hardware
//
#include <hip/hip_runtime.h>
#include <hip/hip_bf16.h>

// ---------------------------------------------------------------------------
// LSTM on MI455X (gfx1250): persistent-kernel recurrence with bf16 WMMA.
//   - 32 blocks x 128 threads (4 waves). Block b owns h-columns [16b, 16b+16)
//     and its i/f/g/o gate columns -> cell update is block/wave-local.
//   - W_hh slice bf16 in LDS (padded, conflict-free) for all 1024 steps.
//   - W_ih pre-converted to bf16 in global memory (1 MB, L2-resident).
//   - h double-buffered in global bf16; each step the whole 64x512 h matrix
//     is DMA'd into LDS by the Tensor Data Mover (tensor_load_to_lds) with
//     hardware row padding (+8 bf16/row) for bank-conflict-free ds reads.
//   - Per-step grid barrier via atomic counter + agent-scope fences.
//   - c (cell state) persists in registers in WMMA C-layout (v8f per wave).
// ---------------------------------------------------------------------------

typedef __attribute__((ext_vector_type(16))) __bf16 v16bf;
typedef __attribute__((ext_vector_type(8)))  __bf16 v8bf;
typedef __attribute__((ext_vector_type(4)))  __bf16 v4bf;
typedef __attribute__((ext_vector_type(8)))  float  v8f;
typedef __attribute__((ext_vector_type(4)))  unsigned v4u_t;
typedef __attribute__((ext_vector_type(8)))  int      v8i_t;
typedef __attribute__((ext_vector_type(4)))  int      v4i_t;

#define B_    64
#define T_    1024
#define D_    256
#define H_    512
#define NBLK  32
#define HC    16          // h columns per block
#define HLD   520         // 512 + 8 pad  -> 4-bank rotation per row

#define HAVE_TDM (__has_builtin(__builtin_amdgcn_tensor_load_to_lds) && \
                  __has_builtin(__builtin_amdgcn_s_wait_tensorcnt))

__device__ __forceinline__ float sigf(float x) {
  return 1.0f / (1.0f + __expf(-x));
}

__device__ __forceinline__ v8bf ld8(const __bf16* p) {
  return *(const v8bf*)p;   // 16B -> global_load_b128 / ds_load_b128
}

__device__ __forceinline__ v16bf mk16(v8bf lo, v8bf hi) {
  return __builtin_shufflevector(lo, hi, 0,1,2,3,4,5,6,7,8,9,10,11,12,13,14,15);
}

// --- prep: generic f32 -> bf16 conversion (used for x and W_ih) ------------
__global__ void convert_bf16_kernel(const float* __restrict__ src,
                                    __bf16* __restrict__ dst, int n4) {
  int i = blockIdx.x * blockDim.x + threadIdx.x;
  if (i < n4) {
    float4 v = ((const float4*)src)[i];
    v4bf o = { (__bf16)v.x, (__bf16)v.y, (__bf16)v.z, (__bf16)v.w };
    ((v4bf*)dst)[i] = o;
  }
}

// --- prep: zero h buffers + barrier counter (every call, for graph replay) -
__global__ void reset_state_kernel(__bf16* __restrict__ hbuf,
                                   unsigned* __restrict__ bar, int n) {
  int i = blockIdx.x * blockDim.x + threadIdx.x;
  if (i < n) hbuf[i] = (__bf16)0.0f;
  if (i == 0) *bar = 0u;
}

// --- persistent recurrence --------------------------------------------------
__global__ void __launch_bounds__(128, 1)
lstm_persistent_kernel(const __bf16* __restrict__ xbf,    // [B, T, D] bf16
                       const __bf16* __restrict__ wihbf,  // [4H, D]   bf16
                       const float*  __restrict__ Whh,    // [4H, H]   f32
                       const float*  __restrict__ bih,
                       const float*  __restrict__ bhh,
                       __bf16*       __restrict__ hbuf,   // 2 * B_*H_ bf16
                       unsigned*     __restrict__ bar,
                       float*        __restrict__ out)    // [T_*B_, H_]
{
  __shared__ __bf16 sWhh[64 * HLD];     // [q*16+n][k], bf16, padded
#if HAVE_TDM
  __shared__ __bf16 sH[B_ * HLD];       // TDM-staged h, padded by TDM
#endif

  const int tid  = threadIdx.x;
  const int wave = tid >> 5;            // 0..3 = M-tile
  const int lane = tid & 31;
  const int half = lane >> 4;
  const int l16  = lane & 15;
  const int hc0  = blockIdx.x * HC;

  // ---- stage W_hh slice into LDS once (f32 -> bf16) ----
  for (int i = tid; i < 64 * H_; i += 128) {
    int row = i / H_, k = i % H_;
    int gcol = (row >> 4) * H_ + hc0 + (row & 15);
    sWhh[row * HLD + k] = (__bf16)Whh[gcol * H_ + k];
  }
  // ---- bias in registers: lane needs gate columns q*H + hc0 + l16 ----
  float bias[4];
#pragma unroll
  for (int q = 0; q < 4; ++q) {
    int gcol = q * H_ + hc0 + l16;
    bias[q] = bih[gcol] + bhh[gcol];
  }
  __syncthreads();

  v8f c = {0.f, 0.f, 0.f, 0.f, 0.f, 0.f, 0.f, 0.f};   // persistent cell state
  unsigned gen = 0;

  const int mtile = wave;
  const int brow  = mtile * 16 + l16;                  // A-matrix row (batch)
  const __bf16* aXbase =
      xbf + (size_t)brow * (T_ * D_) + 8 * half;       // + t*D_ + kc later
  const __bf16* bIhBase =
      wihbf + (size_t)(hc0 + l16) * D_ + 8 * half;     // + q*H_*D_ + kc later
  const int col = hc0 + l16;

#if HAVE_TDM
  // D# group1: data_size=2B, pad_enable, pad_interval=7 (256 DW = 1 row),
  // pad_amount=3 (4 DW = 8 bf16), dims 512x64, tile 512x64, stride 512.
  const v8i_t g1 = { (int)0x07D10000, (int)0x02000000, (int)0x00400000,
                     (int)0x02000000, (int)0x00000040, (int)0x00000200,
                     0, 0 };
  const v4i_t gz = { 0, 0, 0, 0 };
  const unsigned ldsAddr = (unsigned)(size_t)(void*)sH;
#endif

  for (int t = 0; t < T_; ++t) {
    const __bf16* hcur  = hbuf + (size_t)(t & 1) * (B_ * H_);
    __bf16*       hnext = hbuf + (size_t)((t + 1) & 1) * (B_ * H_);

#if HAVE_TDM
    // ---- TDM: DMA h (64x512 bf16) into padded LDS tile, one per block ----
    if (wave == 0) {
      unsigned long long ga = (unsigned long long)(size_t)hcur;
      v4u_t g0 = { 1u,                                   // count=1 (valid D#)
                   ldsAddr,                              // lds_addr
                   (unsigned)(ga & 0xFFFFFFFFu),         // global_addr[31:0]
                   (unsigned)(((ga >> 32) & 0x1FFFFFFu) | (2u << 30)) };
#if __clang_major__ >= 23
      const v8i_t g5z = { 0, 0, 0, 0, 0, 0, 0, 0 };
      __builtin_amdgcn_tensor_load_to_lds(g0, g1, gz, gz, g5z, 0);
#else
      __builtin_amdgcn_tensor_load_to_lds(g0, g1, gz, gz, 0);
#endif
      __builtin_amdgcn_s_wait_tensorcnt(0);
    }
    __syncthreads();
#endif

    // accumulators pre-seeded with bias (same bias for all 8 rows of a lane)
    v8f acc[4];
#pragma unroll
    for (int q = 0; q < 4; ++q) {
      float bq = bias[q];
      acc[q] = (v8f){bq, bq, bq, bq, bq, bq, bq, bq};
    }

    // ---- x @ W_ih^T contribution, K = 256 (A: global x, B: global bf16 W) --
    const __bf16* aX = aXbase + (size_t)t * D_;
#pragma unroll 2
    for (int kc = 0; kc < D_; kc += 32) {
      v16bf a = mk16(ld8(aX + kc), ld8(aX + kc + 16));
#pragma unroll
      for (int q = 0; q < 4; ++q) {
        const __bf16* bp = bIhBase + (size_t)q * (H_ * D_) + kc;
        v16bf bm = mk16(ld8(bp), ld8(bp + 16));
        acc[q] = __builtin_amdgcn_wmma_f32_16x16x32_bf16(
            false, a, false, bm, (short)0, acc[q], false, false);
      }
    }

    // ---- h @ W_hh^T contribution, K = 512 (A: LDS h tile, B: LDS W_hh) ----
#if HAVE_TDM
    const __bf16* aH = &sH[brow * HLD + 8 * half];
#else
    const __bf16* aH = hcur + brow * H_ + 8 * half;
#endif
#pragma unroll 2
    for (int kc = 0; kc < H_; kc += 32) {
      v16bf a = mk16(ld8(aH + kc), ld8(aH + kc + 16));
#pragma unroll
      for (int q = 0; q < 4; ++q) {
        const __bf16* bp = &sWhh[(q * 16 + l16) * HLD + kc + 8 * half];
        v16bf bm = mk16(ld8(bp), ld8(bp + 16));
        acc[q] = __builtin_amdgcn_wmma_f32_16x16x32_bf16(
            false, a, false, bm, (short)0, acc[q], false, false);
      }
    }

    // ---- cell update: element r <-> row b = mtile*16 + r + 8*half ----
    float* outT = out + (size_t)t * (B_ * H_);
#pragma unroll
    for (int r = 0; r < 8; ++r) {
      int b = mtile * 16 + r + 8 * half;
      float iv = acc[0][r], fv = acc[1][r], gv = acc[2][r], ov = acc[3][r];
      float cn = sigf(fv) * c[r] + sigf(iv) * tanhf(gv);
      c[r] = cn;
      float h = sigf(ov) * tanhf(cn);
      hnext[b * H_ + col] = (__bf16)h;
      outT[(size_t)b * H_ + col] = sigf(h);
    }

    // prefetch next timestep's x slice into the cache hierarchy
    if (t + 1 < T_) __builtin_prefetch(aX + D_, 0, 1);

    // ---- device-wide barrier (release h writes, acquire others') ----
    __threadfence();
    __syncthreads();
    if (tid == 0) {
      __hip_atomic_fetch_add(bar, 1u, __ATOMIC_ACQ_REL, __HIP_MEMORY_SCOPE_AGENT);
      unsigned target = (gen + 1u) * (unsigned)NBLK;
      while (__hip_atomic_load(bar, __ATOMIC_ACQUIRE, __HIP_MEMORY_SCOPE_AGENT) < target)
        __builtin_amdgcn_s_sleep(1);
    }
    gen++;
    __syncthreads();
    __threadfence();
  }
}

extern "C" void kernel_launch(void* const* d_in, const int* in_sizes, int n_in,
                              void* d_out, int out_size, void* d_ws, size_t ws_size,
                              hipStream_t stream) {
  const float* x   = (const float*)d_in[0];   // [B, T, D]
  const float* Wih = (const float*)d_in[1];   // [4H, D]
  const float* Whh = (const float*)d_in[2];   // [4H, H]
  const float* bih = (const float*)d_in[3];   // [4H]
  const float* bhh = (const float*)d_in[4];   // [4H]
  float* out = (float*)d_out;                 // [T*B, H]

  char* ws = (char*)d_ws;
  unsigned* bar   = (unsigned*)ws;                                 // 256 B
  __bf16*   hbuf  = (__bf16*)(ws + 256);                           // 128 KB
  __bf16*   xbf   = (__bf16*)(ws + 256 + 2 * B_ * H_ * 2);         // 32 MB
  __bf16*   wihbf = (__bf16*)(ws + 256 + 2 * B_ * H_ * 2
                                 + (size_t)B_ * T_ * D_ * 2);      // 1 MB

  int n4x = (B_ * T_ * D_) / 4;
  int n4w = (4 * H_ * D_) / 4;
  convert_bf16_kernel<<<(n4x + 255) / 256, 256, 0, stream>>>(x, xbf, n4x);
  convert_bf16_kernel<<<(n4w + 255) / 256, 256, 0, stream>>>(Wih, wihbf, n4w);
  reset_state_kernel<<<(2 * B_ * H_ + 255) / 256, 256, 0, stream>>>(
      hbuf, bar, 2 * B_ * H_);
  lstm_persistent_kernel<<<NBLK, 128, 0, stream>>>(
      xbf, wihbf, Whh, bih, bhh, hbuf, bar, out);
}
